// MultiHeadAttention_17463337026273
// MI455X (gfx1250) — matrix operand order repaired and checked by reading, not compiled, not run on hardware
//
#include <hip/hip_runtime.h>

// ---------------- types ----------------
typedef __bf16 bf16_t;
typedef bf16_t v16bf __attribute__((ext_vector_type(16)));
typedef float  v8f   __attribute__((ext_vector_type(8)));
typedef unsigned int v4u __attribute__((ext_vector_type(4)));

union Frag16 {
  v16bf v;
  v4u   q[2];
};

#define HEADS 8
#define HDIM  256      // total model dim
#define EN    32       // neighbors
#define KIN   259      // input features for k/v (H+3)
#define KEXT  288      // logical K extent: 9 * 32
#define KP    296      // LDS row pitch in elements (148 dwords; 148%64=20 -> conflict-free halves)
#define NKS   9        // K steps of 32
#define DH    32       // per-head dim

#define WPACK_ELEMS (16 * NKS * 32 * 16)          // per tensor: 73728 bf16
#define WPACK_BYTES (2 * WPACK_ELEMS * 2)         // 294912 B
#define WT_BYTES    (HDIM * HDIM * 4)             // 262144 B

// round-to-nearest-even f32 -> bf16 bits
__device__ __forceinline__ unsigned short f2bf(float f) {
  unsigned int u = __float_as_uint(f);
  u += 0x7FFFu + ((u >> 16) & 1u);
  return (unsigned short)(u >> 16);
}

// -------------------------------------------------------------------------
// Pack Wk and Wv (f32, [256][259] row-major) into bf16 WMMA B-fragment order:
//   [tensor(2)][ntile(16)][kstep(9)][lane(32)][j(16)]
// B 32x16 bf16 packing used here (contiguous K, same as A loads): lanes 0-15: col=lane, K=k0+j; lanes 16-31:
// col=lane-16, K=k0+16+j.  K >= 259 zero-padded.
// -------------------------------------------------------------------------
__global__ void pack_w_kernel(const float* __restrict__ Wk,
                              const float* __restrict__ Wv,
                              unsigned short* __restrict__ outp) {
  int tid  = blockIdx.x * blockDim.x + threadIdx.x;   // 0 .. 9215
  int lane = tid & 31;
  int ks   = (tid >> 5) % NKS;
  int nt   = ((tid >> 5) / NKS) % 16;
  int tsel = tid / (32 * NKS * 16);
  const float* W = (tsel == 0) ? Wk : Wv;
  int col   = lane & 15;
  int kbase = ks * 32 + ((lane < 16) ? 0 : 16);
  unsigned short* o = outp + (size_t)tid * 16;
#pragma unroll
  for (int j = 0; j < 16; ++j) {
    int k = kbase + j;
    float val = (k < KIN) ? W[(nt * 16 + col) * KIN + k] : 0.0f;
    o[j] = f2bf(val);
  }
}

// Transpose Wq and Wo (f32 256x256) so GEMV reads are lane-coalesced.
__global__ void transpose_w_kernel(const float* __restrict__ Wq,
                                   const float* __restrict__ Wo,
                                   float* __restrict__ wqT,
                                   float* __restrict__ woT) {
  int tid = blockIdx.x * blockDim.x + threadIdx.x;    // 0 .. 2*65536-1
  int o = tid & (HDIM - 1);
  int c = (tid >> 8) & (HDIM - 1);
  int tsel = tid >> 16;
  if (tsel == 0) wqT[c * HDIM + o] = Wq[o * HDIM + c];
  else           woT[c * HDIM + o] = Wo[o * HDIM + c];
}

// -------------------------------------------------------------------------
// 36-WMMA projection of one 32x259 LDS tile against one head's packed weight
// columns (two 16-wide N tiles), results staged to per-head LDS (pitch 33).
// -------------------------------------------------------------------------
__device__ __forceinline__ void proj_head(const unsigned short* __restrict__ ldsrc,
                                          const unsigned short* __restrict__ wp,
                                          int h, int lane,
                                          float* __restrict__ stage) {
  const int arow = lane & 15;
  const int ac0  = (lane < 16) ? 0 : 16;  // A chunk element offsets: contiguous K=16*half+j (matches B pack)
  const int ac1  = ac0 + 8;

  v8f c00 = {0.f,0.f,0.f,0.f,0.f,0.f,0.f,0.f};
  v8f c01 = c00, c10 = c00, c11 = c00;

#pragma unroll
  for (int ks = 0; ks < NKS; ++ks) {
    const int k0 = ks * 32;
    Frag16 a0, a1, b0, b1;
    a0.q[0] = *(const v4u*)&ldsrc[arow * KP + k0 + ac0];
    a0.q[1] = *(const v4u*)&ldsrc[arow * KP + k0 + ac1];
    a1.q[0] = *(const v4u*)&ldsrc[(16 + arow) * KP + k0 + ac0];
    a1.q[1] = *(const v4u*)&ldsrc[(16 + arow) * KP + k0 + ac1];
    const unsigned short* bp0 = wp + ((((2 * h)     * NKS) + ks) * 32 + lane) * 16;
    const unsigned short* bp1 = wp + ((((2 * h + 1) * NKS) + ks) * 32 + lane) * 16;
    b0.q[0] = *(const v4u*)(bp0);
    b0.q[1] = *(const v4u*)(bp0 + 8);
    b1.q[0] = *(const v4u*)(bp1);
    b1.q[1] = *(const v4u*)(bp1 + 8);

    c00 = __builtin_amdgcn_wmma_f32_16x16x32_bf16(false, a0.v, false, b0.v, (short)0, c00, false, false);
    c01 = __builtin_amdgcn_wmma_f32_16x16x32_bf16(false, a0.v, false, b1.v, (short)0, c01, false, false);
    c10 = __builtin_amdgcn_wmma_f32_16x16x32_bf16(false, a1.v, false, b0.v, (short)0, c10, false, false);
    c11 = __builtin_amdgcn_wmma_f32_16x16x32_bf16(false, a1.v, false, b1.v, (short)0, c11, false, false);
  }

  // C/D 16x16 f32 layout: VGPR r: lanes 0-15 -> M=r, N=lane; lanes 16-31 -> M=8+r
  const int eoff = (lane >> 4) << 3;   // 0 or 8
  const int dd0  = lane & 15;
#pragma unroll
  for (int r = 0; r < 8; ++r) {
    stage[(r + eoff) * 33 + dd0]           = c00[r];
    stage[(r + eoff) * 33 + 16 + dd0]      = c01[r];
    stage[(16 + r + eoff) * 33 + dd0]      = c10[r];
    stage[(16 + r + eoff) * 33 + 16 + dd0] = c11[r];
  }
}

// -------------------------------------------------------------------------
// Fused per-node multi-head neighbor attention. One block (8 waves) per node;
// wave h owns head h. LDS ~74 KB -> 4 workgroups per 320 KB WGP.
// -------------------------------------------------------------------------
__global__ void __launch_bounds__(256)
attn_kernel(const float* __restrict__ qg,
            const float* __restrict__ kg,
            const float* __restrict__ vg,
            const float* __restrict__ wqT,
            const float* __restrict__ woT,
            const unsigned short* __restrict__ wpack,
            float* __restrict__ outg, int nnodes) {
  __shared__ alignas(16) unsigned short lds_k[EN * KP];   // 18.5 KB bf16
  __shared__ alignas(16) unsigned short lds_v[EN * KP];   // 18.5 KB bf16
  __shared__ float lds_q[HDIM];
  __shared__ float lds_qp[HDIM];
  __shared__ float lds_st[HEADS][EN * 33];                // 33 KB, kp then vp
  __shared__ float lds_att[HEADS][EN];
  __shared__ float lds_res[HDIM];

  const int t    = threadIdx.x;
  const int lane = t & 31;
  const int h    = t >> 5;
  const int n    = blockIdx.x;
  if (n >= nnodes) return;

  // ---- cooperative load: q[n] (f32->LDS), k[n]/v[n] (f32 -> bf16 LDS, pad K) ----
  const float* kb = kg + (size_t)n * EN * KIN;
  const float* vb = vg + (size_t)n * EN * KIN;
  lds_q[t] = qg[(size_t)n * HDIM + t];
#pragma unroll 4
  for (int r = 0; r < EN; ++r) {
    lds_k[r * KP + t] = f2bf(kb[r * KIN + t]);
    lds_v[r * KP + t] = f2bf(vb[r * KIN + t]);
    if (t < (KP - 256)) {                    // cols 256..295 (tail + zero pad)
      int c = 256 + t;
      float kv = (c < KIN) ? kb[r * KIN + c] : 0.0f;
      float vv = (c < KIN) ? vb[r * KIN + c] : 0.0f;
      lds_k[r * KP + c] = f2bf(kv);
      lds_v[r * KP + c] = f2bf(vv);
    }
  }
  __syncthreads();

  // ---- qp[t] = sum_c Wq[t][c]*q[c], coalesced via WqT (head h's qp is wave-local) ----
  {
    float acc = 0.0f;
#pragma unroll 8
    for (int c = 0; c < HDIM; ++c) acc += wqT[c * HDIM + t] * lds_q[c];
    lds_qp[t] = acc;
  }

  // ---- phase 1: kp = k * Wk^T (head cols), stage, then simi + softmax ----
  proj_head(lds_k, wpack, h, lane, &lds_st[h][0]);
  {
    float simi = 0.0f;
#pragma unroll 8
    for (int dd = 0; dd < DH; ++dd)
      simi += lds_st[h][lane * 33 + dd] * lds_qp[h * DH + dd];
    simi *= 0.17677669529663687f;            // 1/sqrt(32)

    float m = simi;
#pragma unroll
    for (int mk = 16; mk >= 1; mk >>= 1)
      m = fmaxf(m, __shfl_xor(m, mk, 32));
    float p = __expf(simi - m);
    float s = p;
#pragma unroll
    for (int mk = 16; mk >= 1; mk >>= 1)
      s += __shfl_xor(s, mk, 32);
    lds_att[h][lane] = p / s;
  }

  // ---- phase 2: vp = v * Wv^T, stage (same buffer), att-weighted sum ----
  proj_head(lds_v, wpack + WPACK_ELEMS, h, lane, &lds_st[h][0]);
  {
    float o = 0.0f;
#pragma unroll 8
    for (int e = 0; e < EN; ++e)
      o += lds_st[h][e * 33 + lane] * lds_att[h][e];
    lds_res[h * DH + lane] = o;
  }
  __syncthreads();

  // ---- final projection: out[o] = sum_d Wo[o][d]*res[d], coalesced via WoT ----
  {
    float acc = 0.0f;
#pragma unroll 8
    for (int c = 0; c < HDIM; ++c) acc += woT[c * HDIM + t] * lds_res[c];
    outg[(size_t)n * HDIM + t] = acc;
  }
}

// -------------------------------------------------------------------------
extern "C" void kernel_launch(void* const* d_in, const int* in_sizes, int n_in,
                              void* d_out, int out_size, void* d_ws, size_t ws_size,
                              hipStream_t stream) {
  (void)n_in; (void)out_size; (void)ws_size;
  const float* q  = (const float*)d_in[0];
  const float* k  = (const float*)d_in[1];
  const float* v  = (const float*)d_in[2];
  const float* Wq = (const float*)d_in[3];
  const float* Wk = (const float*)d_in[4];
  const float* Wv = (const float*)d_in[5];
  const float* Wo = (const float*)d_in[6];
  float* out = (float*)d_out;

  unsigned short* wpack = (unsigned short*)d_ws;                 // 288 KB bf16 fragments
  float* wqT = (float*)((char*)d_ws + WPACK_BYTES);              // 256 KB
  float* woT = (float*)((char*)d_ws + WPACK_BYTES + WT_BYTES);   // 256 KB

  const int nnodes = in_sizes[0] / HDIM;   // 10000
  if (nnodes <= 0) return;

  pack_w_kernel<<<36, 256, 0, stream>>>(Wk, Wv, wpack);
  transpose_w_kernel<<<(2 * HDIM * HDIM) / 256, 256, 0, stream>>>(Wq, Wo, wqT, woT);
  attn_kernel<<<nnodes, 256, 0, stream>>>(q, k, v, wqT, woT, wpack, out, nnodes);
}
